// GINNet_73486890435219
// MI455X (gfx1250) — compile-verified, matrix operand order checked by reading, not run on hardware
//
#include <hip/hip_runtime.h>
#include <hip/hip_bf16.h>
#include <stdint.h>

typedef _Float16 half_t;
typedef __attribute__((ext_vector_type(16))) _Float16 v16h;
typedef __attribute__((ext_vector_type(8)))  _Float16 v8h;
typedef __attribute__((ext_vector_type(8)))  float    v8f;
typedef __attribute__((ext_vector_type(4)))  int      v4i;

#define NN 100000     // nodes
#define NE1 250000
#define NE2 250000
#define NET 500000    // total edges
#define NG 1024       // graphs
#define GDIM 108

#if __has_builtin(__builtin_amdgcn_global_load_async_to_lds_b128) && \
    __has_builtin(__builtin_amdgcn_s_wait_asynccnt)
#define USE_ASYNC_LDS 1
#else
#define USE_ASYNC_LDS 0
#endif

// ---------------------------------------------------------------------------
// Weight packing: rewrite f32 (k_real x n_real, row-major) weights into the
// CDNA5 WMMA 16-bit B-matrix (32x16 per k-tile) per-lane fragment order so the
// GEMM reads B with one coalesced 32B load per lane. Zero-padded in both K and
// N, so the GEMM can run branch-free over padded tiles.
// frag f = (ntile*Ktiles + ktile); lane l: col = nt*16+(l&15),
// k offsets = base + j + (j>=8 ? 8 : 0), base = (l>=16 ? 8 : 0).
// ---------------------------------------------------------------------------
__global__ void pack_b(const float* __restrict__ W, half_t* __restrict__ Bp,
                       int Ktiles, int Ntiles, int k_real, int n_real)
{
    int idx = blockIdx.x * blockDim.x + threadIdx.x;
    int total = Ktiles * Ntiles * 32;
    if (idx >= total) return;
    int lane = idx & 31;
    int frag = idx >> 5;
    int kt = frag % Ktiles;
    int nt = frag / Ktiles;
    int col = nt * 16 + (lane & 15);
    int kb  = kt * 32 + ((lane >> 4) << 3);
    v16h o;
#pragma unroll
    for (int j = 0; j < 16; ++j) {
        int k = kb + j + ((j >= 8) ? 8 : 0);
        float v = (k < k_real && col < n_real) ? W[(size_t)k * n_real + col] : 0.0f;
        o[j] = (half_t)v;
    }
    *(v16h*)(Bp + (size_t)idx * 16) = o;
}

// ---------------------------------------------------------------------------
// WMMA GEMM: C(MxN,f32-acc) = A(MxK,f16) * B(packed,f16) + bias, optional relu.
// 256 threads = 8 waves; wave w owns m-tile (blockIdx.x*8+w) and NT n-tiles at
// blockIdx.y*NT*16. The block's B panel (contiguous in the packed buffer,
// Ktiles*NT KB) is staged once into LDS via the gfx1250 async global->LDS path
// and shared by all 8 waves; A streams from global with prefetch.
// Requires: M % 16 == 0, Ntiles % NT == 0 (enforced by padded packing).
// ---------------------------------------------------------------------------
template <bool RELU, int NT>
__global__ __launch_bounds__(256) void gemm_wmma_f16(
    const half_t* __restrict__ A, const half_t* __restrict__ Bp,
    const float* __restrict__ bias,
    half_t* __restrict__ C16, int ldc16,
    float* __restrict__ C32, int ldc32,
    int Mtiles, int Ktiles, int lda, int n_real)
{
    extern __shared__ char lds_raw[];
    half_t* lds_b = (half_t*)lds_raw;

    const int tid = threadIdx.x;
    const int wave = tid >> 5;
    const int lane = tid & 31;
    const int mt = blockIdx.x * 8 + wave;
    const int nt0 = blockIdx.y * NT;

    // ---- stage the block's B panel into LDS (all threads participate) ----
    const size_t panel_halves = (size_t)Ktiles * NT * 512;   // NT*Ktiles frags * 32 lanes * 16 halves
    const half_t* gpanel = Bp + (size_t)nt0 * Ktiles * 512;
#if USE_ASYNC_LDS
    for (size_t off = (size_t)tid * 8; off < panel_halves; off += 256 * 8) {
        __builtin_amdgcn_global_load_async_to_lds_b128(
            (__attribute__((address_space(1))) v4i*)(gpanel + off),
            (__attribute__((address_space(3))) v4i*)(lds_b + off), 0, 0);
    }
    __builtin_amdgcn_s_wait_asynccnt(0);
#else
    for (size_t off = (size_t)tid * 8; off < panel_halves; off += 256 * 8) {
        *(v8h*)(lds_b + off) = *(const v8h*)(gpanel + off);
    }
#endif
    __syncthreads();

    if (mt >= Mtiles) return;

    const int row = mt * 16 + (lane & 15);
    const int kshift = (lane >> 4) << 3;
    const half_t* arow = A + (size_t)row * lda + kshift;
    const int bstep = Ktiles * 512;   // halves between n-tile frags within panel

    v8f acc[NT];
#pragma unroll
    for (int j = 0; j < NT; ++j) acc[j] = (v8f){};

    for (int kt = 0; kt < Ktiles; ++kt) {
        const v8h* ap = (const v8h*)(arow + kt * 32);
        v8h alo = ap[0];
        v8h ahi = ap[2];
        v16h a;
#pragma unroll
        for (int i = 0; i < 8; ++i) { a[i] = alo[i]; a[8 + i] = ahi[i]; }
        const half_t* bbase = lds_b + ((size_t)kt * 32 + lane) * 16;
#pragma unroll
        for (int j = 0; j < NT; ++j) {
            v16h b = *(const v16h*)(bbase + j * bstep);
            acc[j] = __builtin_amdgcn_wmma_f32_16x16x32_f16(false, a, false, b,
                                                            (short)0, acc[j], false, false);
        }
        // speculative prefetch of the next A k-slice (drops silently if OOB)
        __builtin_prefetch(arow + (kt + 1) * 32, 0, 1);
    }
    // C layout: vgpr r -> row mt*16 + r + (lane>=16 ? 8 : 0), col = ntile*16 + (lane&15)
    const int rbase = mt * 16 + ((lane >> 4) << 3);
    const int cgrp = lane & 15;
#pragma unroll
    for (int j = 0; j < NT; ++j) {
        int col = (nt0 + j) * 16 + cgrp;
        float bv = (bias != nullptr && col < n_real) ? bias[col] : 0.0f;
#pragma unroll
        for (int r = 0; r < 8; ++r) {
            float vv = acc[j][r] + bv;
            if (RELU) vv = fmaxf(vv, 0.0f);
            size_t ro = (size_t)(rbase + r);
            if (C16) C16[ro * ldc16 + col] = (half_t)vv;
            if (C32) C32[ro * ldc32 + col] = vv;
        }
    }
}

// ---------------------------------------------------------------------------
// Feature construction
// ---------------------------------------------------------------------------
struct EmbPtrs9 { const float* p[9]; };
struct EmbPtrs5 { const float* p[5]; };

__global__ void node_feats(const float* __restrict__ x, const float* __restrict__ pos,
                           EmbPtrs9 emb, float* __restrict__ xf32, half_t* __restrict__ xf16)
{
    int i = blockIdx.x * blockDim.x + threadIdx.x;
    if (i >= NN) return;
    const int dims[9] = {5, 2, 6, 3, 5, 4, 6, 1, 1};
    float f[37];
    int o = 0;
#pragma unroll
    for (int c = 0; c < 9; ++c) {
        int id = (int)x[(size_t)i * 13 + c];
        const float* t = emb.p[c] + (size_t)id * dims[c];
        for (int j = 0; j < dims[c]; ++j) f[o++] = t[j];
    }
    f[o++] = x[(size_t)i * 13 + 9];
    f[o++] = pos[(size_t)i * 3 + 0] * 0.1f;
    f[o++] = pos[(size_t)i * 3 + 1] * 0.1f;
    f[o++] = pos[(size_t)i * 3 + 2] * 0.1f;
#pragma unroll
    for (int d = 0; d < 37; ++d) {
        xf32[(size_t)i * 64 + d] = f[d];
        xf16[(size_t)i * 64 + d] = (half_t)f[d];
    }
}

__global__ void edge_feats(const float* __restrict__ ea1, const float* __restrict__ ea2,
                           EmbPtrs5 emb, half_t* __restrict__ ea16)
{
    int e = blockIdx.x * blockDim.x + threadIdx.x;
    if (e >= NET) return;
    const float* row = (e < NE1) ? (ea1 + (size_t)e * 6) : (ea2 + (size_t)(e - NE1) * 6);
    const int dims[5] = {3, 2, 3, 1, 1};
    float f[11];
    int o = 0;
#pragma unroll
    for (int c = 0; c < 5; ++c) {
        int id = (int)row[c];
        const float* t = emb.p[c] + (size_t)id * dims[c];
        for (int j = 0; j < dims[c]; ++j) f[o++] = t[j];
    }
    f[o++] = row[5] * 0.1f;
#pragma unroll
    for (int d = 0; d < 11; ++d) ea16[(size_t)e * 32 + d] = (half_t)f[d];
}

// ---------------------------------------------------------------------------
// Graph ops
// ---------------------------------------------------------------------------
__global__ void ba_msg_agg(const float* __restrict__ H, const float* __restrict__ EW,
                           const int* __restrict__ ei, float* __restrict__ AGG, int ne)
{
    int e = blockIdx.x, d = threadIdx.x;
    if (d >= GDIM) return;
    int src = ei[e], dst = ei[ne + e];
    float v = H[(size_t)src * 128 + d] + EW[(size_t)e * 128 + d];
    atomicAdd(&AGG[(size_t)dst * 128 + d], fmaxf(v, 0.0f));
}

__global__ void scatter_add(const float* __restrict__ X, const int* __restrict__ ei,
                            float* __restrict__ AGG, int ne, int ld, int cols)
{
    int e = blockIdx.x, d = threadIdx.x;
    if (d >= cols) return;
    int src = ei[e], dst = ei[ne + e];
    atomicAdd(&AGG[(size_t)dst * ld + d], X[(size_t)src * ld + d]);
}

__global__ void add_to_h16(const float* __restrict__ A, const float* __restrict__ B,
                           half_t* __restrict__ O, int ld, int cols)
{
    int r = blockIdx.x, d = threadIdx.x;
    if (d >= cols) return;
    O[(size_t)r * ld + d] = (half_t)(A[(size_t)r * ld + d] + B[(size_t)r * ld + d]);
}

__global__ void attn_score(const float* __restrict__ T1, const float* __restrict__ T2,
                           const float* __restrict__ v, float* __restrict__ alpha)
{
    int n = blockIdx.x * blockDim.x + threadIdx.x;
    if (n >= NN) return;
    float s1 = 0.0f, s2 = 0.0f;
    for (int d = 0; d < GDIM; ++d) {
        float vd = v[d];
        s1 += tanhf(T1[(size_t)n * 128 + d]) * vd;
        s2 += tanhf(T2[(size_t)n * 128 + d]) * vd;
    }
    float m = fmaxf(s1, s2);
    float e1 = expf(s1 - m), e2 = expf(s2 - m);
    alpha[n] = e1 / (e1 + e2);
}

__global__ void attn_combine(const float* __restrict__ H1, const float* __restrict__ H2,
                             const float* __restrict__ alpha, float* __restrict__ O)
{
    int n = blockIdx.x, d = threadIdx.x;
    if (d >= GDIM) return;
    float a = alpha[n];
    float y = a * H1[(size_t)n * 128 + d] + (1.0f - a) * H2[(size_t)n * 128 + d];
    O[(size_t)n * 128 + d] = fmaxf(y, 0.0f);
}

// ---------------------------------------------------------------------------
// BatchNorm (two pass: atomic column stats, then normalize)
// ---------------------------------------------------------------------------
template <typename T>
__global__ void bn_stats_t(const T* __restrict__ X, float* __restrict__ st,
                           int rows, int cols, int ld, int chunk)
{
    int idx = blockIdx.x * blockDim.x + threadIdx.x;
    int nchunks = (rows + chunk - 1) / chunk;
    if (idx >= nchunks * cols) return;
    int c = idx % cols;
    int r0 = (idx / cols) * chunk;
    int r1 = r0 + chunk; if (r1 > rows) r1 = rows;
    float s = 0.0f, s2 = 0.0f;
    for (int r = r0; r < r1; ++r) {
        float v = (float)X[(size_t)r * ld + c];
        s += v; s2 += v * v;
    }
    atomicAdd(&st[c], s);
    atomicAdd(&st[cols + c], s2);
}

__global__ void bn_apply(const float* __restrict__ X, const float* __restrict__ st,
                         const float* __restrict__ g, const float* __restrict__ b,
                         float* __restrict__ Y32, half_t* __restrict__ Y16,
                         int cols, int ld, float invN)
{
    int r = blockIdx.x, c = threadIdx.x;
    if (c >= cols) return;
    float mean = st[c] * invN;
    float var = st[cols + c] * invN - mean * mean;
    float y = (X[(size_t)r * ld + c] - mean) * rsqrtf(var + 1e-5f) * g[c] + b[c];
    if (Y32) Y32[(size_t)r * ld + c] = y;
    if (Y16) Y16[(size_t)r * ld + c] = (half_t)y;
}

__global__ void bn3_apply_seg(const half_t* __restrict__ X, const float* __restrict__ st,
                              const float* __restrict__ g, const float* __restrict__ b,
                              const int* __restrict__ batch, float* __restrict__ out,
                              float invN)
{
    int e = blockIdx.x, c = threadIdx.x;  // block = 256
    float mean = st[c] * invN;
    float var = st[256 + c] * invN - mean * mean;
    float y = ((float)X[(size_t)e * 256 + c] - mean) * rsqrtf(var + 1e-5f) * g[c] + b[c];
    atomicAdd(&out[(size_t)batch[e] * 256 + c], y);
}

__global__ void build_xe(const float* __restrict__ H, const half_t* __restrict__ Ef,
                         const int* __restrict__ ei1, const int* __restrict__ ei2,
                         half_t* __restrict__ XE)
{
    int e = blockIdx.x * blockDim.x + threadIdx.x;
    if (e >= NET) return;
    int src, dst;
    if (e < NE1) { src = ei1[e]; dst = ei1[NE1 + e]; }
    else         { src = ei2[e - NE1]; dst = ei2[NE2 + e - NE1]; }
    half_t* o = XE + (size_t)e * 256;
    const float* hs = H + (size_t)src * 128;
    const float* hd = H + (size_t)dst * 128;
    for (int d = 0; d < GDIM; ++d) o[d] = (half_t)hs[d];
    for (int d = 0; d < GDIM; ++d) o[GDIM + d] = (half_t)hd[d];
    const half_t* ee = Ef + (size_t)e * 64;
    for (int d = 0; d < 40; ++d) o[216 + d] = ee[d];
}

__global__ void seg_out(const float* __restrict__ O, const int* __restrict__ batch,
                        float* __restrict__ out)
{
    int e = blockIdx.x * blockDim.x + threadIdx.x;
    if (e >= NET) return;
    atomicAdd(&out[batch[e]], O[(size_t)e * 64]);
}

// ---------------------------------------------------------------------------
// Host orchestration
// ---------------------------------------------------------------------------
static inline void gemm(hipStream_t s, const half_t* A, const half_t* Bp, const float* bias,
                        half_t* C16, int ldc16, float* C32, int ldc32,
                        int M, int Kpad, int Npad, int lda, int n_real, bool relu, int NT)
{
    int Mtiles = M / 16, Ktiles = Kpad / 32, Ntiles = Npad / 16;
    dim3 grid((Mtiles + 7) / 8, Ntiles / NT);
    size_t shmem = (size_t)Ktiles * NT * 1024;   // B panel bytes
    if (NT == 8) {
        if (relu) gemm_wmma_f16<true, 8><<<grid, 256, shmem, s>>>(A, Bp, bias, C16, ldc16, C32, ldc32, Mtiles, Ktiles, lda, n_real);
        else      gemm_wmma_f16<false, 8><<<grid, 256, shmem, s>>>(A, Bp, bias, C16, ldc16, C32, ldc32, Mtiles, Ktiles, lda, n_real);
    } else {
        if (relu) gemm_wmma_f16<true, 4><<<grid, 256, shmem, s>>>(A, Bp, bias, C16, ldc16, C32, ldc32, Mtiles, Ktiles, lda, n_real);
        else      gemm_wmma_f16<false, 4><<<grid, 256, shmem, s>>>(A, Bp, bias, C16, ldc16, C32, ldc32, Mtiles, Ktiles, lda, n_real);
    }
}

static inline void packw(hipStream_t s, const float* W, half_t* Bp,
                         int Kpad, int Npad, int k_real, int n_real)
{
    int total = (Kpad / 32) * (Npad / 16) * 32;
    pack_b<<<(total + 255) / 256, 256, 0, s>>>(W, Bp, Kpad / 32, Npad / 16, k_real, n_real);
}

extern "C" void kernel_launch(void* const* d_in, const int* in_sizes, int n_in,
                              void* d_out, int out_size, void* d_ws, size_t ws_size,
                              hipStream_t stream)
{
    // ---- inputs (setup_inputs order, params flattened in sorted pytree order)
    const float* x      = (const float*)d_in[0];
    const float* pos    = (const float*)d_in[1];
    const int*   ei1    = (const int*)d_in[2];
    const int*   ei2    = (const int*)d_in[3];
    const float* ea1    = (const float*)d_in[4];
    const float* ea2    = (const float*)d_in[5];
    const int*   ebatch = (const int*)d_in[6];
    #define FP(i) ((const float*)d_in[i])
    const float *attn_Wb = FP(7), *attn_Ww = FP(8), *attn_v = FP(9);
    const float *bn1_b = FP(10), *bn1_g = FP(11);
    const float *bn2_b = FP(12), *bn2_g = FP(13);
    const float *bn3_b = FP(14), *bn3_g = FP(15);
    const float *c1e_b = FP(16), *c1e_w = FP(17), *c1l_b = FP(18), *c1l_w = FP(19);
    const float *c2e_b = FP(20), *c2e_w = FP(21), *c2l_b = FP(22), *c2l_w = FP(23);
    const float *el0_b = FP(24), *el0_w = FP(25), *el1_b = FP(26), *el1_w = FP(27);
    const float *el2_b = FP(28), *el2_w = FP(29);
    EmbPtrs5 eemb; for (int i = 0; i < 5; ++i) eemb.p[i] = FP(30 + i);
    EmbPtrs9 emb;  for (int i = 0; i < 9; ++i) emb.p[i]  = FP(35 + i);
    const float *l0_b = FP(44), *l0_w = FP(45), *l1_b = FP(46), *l1_w = FP(47);
    const float *l2_b = FP(48), *l2_w = FP(49), *l3_b = FP(50), *l3_w = FP(51);
    const float *le0_b = FP(52), *le0_w = FP(53), *le1_b = FP(54), *le1_w = FP(55);
    const float *le2_b = FP(56), *le2_w = FP(57);
    const float *xn1_0b = FP(58), *xn1_0w = FP(59), *xn1_1b = FP(60), *xn1_1w = FP(61);
    const float *xn1a_0b = FP(62), *xn1a_0w = FP(63), *xn1a_1b = FP(64), *xn1a_1w = FP(65);
    const float *xn2_0b = FP(66), *xn2_0w = FP(67), *xn2_1b = FP(68), *xn2_1w = FP(69);
    float* outp = (float*)d_out;   // [0,1024) = out ; [1024, 1024+1024*256) = xe_global

    // zero workspace (atomic accumulators + f16 K-padding) and output
    (void)hipMemsetAsync(d_ws, 0, ws_size, stream);
    (void)hipMemsetAsync(d_out, 0, (size_t)out_size * sizeof(float), stream);

    // ---- bump allocator
    uintptr_t cur = ((uintptr_t)d_ws + 255) & ~(uintptr_t)255;
    auto alloc = [&](size_t bytes) -> void* {
        void* r = (void*)cur;
        cur += (bytes + 255) & ~(size_t)255;
        return r;
    };
    auto ah = [&](size_t n) { return (half_t*)alloc(n * sizeof(half_t)); };
    auto af = [&](size_t n) { return (float*)alloc(n * sizeof(float)); };

    // ---- persistent region (survives the edge phase)
    // packed weights: Ktiles*Ntiles*1024 bytes each (halves = frags*32lanes*16)
    half_t* P_c1l  = ah((size_t)2 * 8 * 512);   // K64  N128
    half_t* P_c1e  = ah((size_t)1 * 8 * 512);   // K32  N128
    half_t* P_c2l  = ah((size_t)4 * 8 * 512);   // K128 N128
    half_t* P_c2e  = ah((size_t)1 * 8 * 512);
    half_t* P_x10  = ah((size_t)2 * 8 * 512);
    half_t* P_x11  = ah((size_t)4 * 8 * 512);
    half_t* P_xa0  = ah((size_t)4 * 8 * 512);
    half_t* P_xa1  = ah((size_t)4 * 8 * 512);
    half_t* P_x20  = ah((size_t)4 * 8 * 512);
    half_t* P_x21  = ah((size_t)4 * 8 * 512);
    half_t* P_attn = ah((size_t)4 * 8 * 512);
    half_t* P_el0  = ah((size_t)1 * 4 * 512);   // K32  N64
    half_t* P_el1  = ah((size_t)1 * 4 * 512);   // K32  N64
    half_t* P_el2  = ah((size_t)2 * 4 * 512);   // K64  N64
    half_t* P_le0  = ah((size_t)8 * 32 * 512);  // K256 N512
    half_t* P_le1  = ah((size_t)16 * 32 * 512); // K512 N512
    half_t* P_le2  = ah((size_t)16 * 16 * 512); // K512 N256
    half_t* P_l0   = ah((size_t)8 * 32 * 512);
    half_t* P_l1   = ah((size_t)16 * 32 * 512);
    half_t* P_l2   = ah((size_t)16 * 16 * 512);
    half_t* P_l3   = ah((size_t)8 * 4 * 512);   // K256 N64
    half_t* EA16   = ah((size_t)NET * 32);
    float*  HFIN32 = af((size_t)NN * 128);
    float*  ALPHA  = af(NN);
    float*  ST1 = af(256), *ST2 = af(256), *ST3 = af(512);

    uintptr_t scratch_mark = cur;

    // ---- node-phase scratch
    float*  XF32  = af((size_t)NN * 64);
    half_t* XF16  = ah((size_t)NN * 64);
    float*  H32   = af((size_t)NN * 128);
    float*  EW32  = af((size_t)NE1 * 128);
    float*  AGG1  = af((size_t)NN * 128);   // conv1 message agg (pre-zeroed)
    float*  AGG1D = af((size_t)NN * 128);   // conv2 message agg
    float*  AGGX  = af((size_t)NN * 64);    // gin1 agg (37 dims)
    float*  AGGH  = af((size_t)NN * 128);   // gin2 agg (108 dims)
    half_t* HS16  = ah((size_t)NN * 128);
    half_t* M16   = ah((size_t)NN * 128);
    half_t* XA16  = ah((size_t)NN * 64);
    half_t* XAH16 = ah((size_t)NN * 128);
    float*  H1_32 = af((size_t)NN * 128);  half_t* H1_16 = ah((size_t)NN * 128);
    float*  H2_32 = af((size_t)NN * 128);  half_t* H2_16 = ah((size_t)NN * 128);
    float*  T1_32 = af((size_t)NN * 128);
    float*  T2_32 = af((size_t)NN * 128);
    float*  HATT  = af((size_t)NN * 128);
    float*  HBN32 = af((size_t)NN * 128);  half_t* HBN16 = ah((size_t)NN * 128);

    // ---- pack all weights (tiny; re-done every call, deterministic)
    packw(stream, c1l_w,  P_c1l, 64, 128, 37, 108);
    packw(stream, c1e_w,  P_c1e, 32, 128, 11, 108);
    packw(stream, c2l_w,  P_c2l, 128, 128, 108, 108);
    packw(stream, c2e_w,  P_c2e, 32, 128, 11, 108);
    packw(stream, xn1_0w, P_x10, 64, 128, 37, 108);
    packw(stream, xn1_1w, P_x11, 128, 128, 108, 108);
    packw(stream, xn1a_0w,P_xa0, 128, 128, 108, 108);
    packw(stream, xn1a_1w,P_xa1, 128, 128, 108, 108);
    packw(stream, xn2_0w, P_x20, 128, 128, 108, 108);
    packw(stream, xn2_1w, P_x21, 128, 128, 108, 108);
    packw(stream, attn_Ww,P_attn,128, 128, 108, 108);
    packw(stream, el0_w,  P_el0, 32, 64, 11, 22);
    packw(stream, el1_w,  P_el1, 32, 64, 22, 40);
    packw(stream, el2_w,  P_el2, 64, 64, 40, 40);
    packw(stream, le0_w,  P_le0, 256, 512, 256, 512);
    packw(stream, le1_w,  P_le1, 512, 512, 512, 512);
    packw(stream, le2_w,  P_le2, 512, 256, 512, 256);
    packw(stream, l0_w,   P_l0,  256, 512, 256, 512);
    packw(stream, l1_w,   P_l1,  512, 512, 512, 512);
    packw(stream, l2_w,   P_l2,  512, 256, 512, 256);
    packw(stream, l3_w,   P_l3,  256, 64, 256, 1);

    // ---- features
    node_feats<<<(NN + 127) / 128, 128, 0, stream>>>(x, pos, emb, XF32, XF16);
    edge_feats<<<(NET + 127) / 128, 128, 0, stream>>>(ea1, ea2, eemb, EA16);

    // ================= conv1 (ba_gin) =================
    gemm(stream, XF16, P_c1l, c1l_b, nullptr, 0, H32, 128, NN, 64, 128, 64, 108, false, 8);
    gemm(stream, EA16, P_c1e, c1e_b, nullptr, 0, EW32, 128, NE1, 32, 128, 32, 108, false, 8);
    ba_msg_agg<<<NE1, 128, 0, stream>>>(H32, EW32, ei1, AGG1, NE1);
    add_to_h16<<<NN, 128, 0, stream>>>(H32, AGG1, HS16, 128, 108);
    gemm(stream, HS16, P_xa0, xn1a_0b, M16, 128, nullptr, 0, NN, 128, 128, 128, 108, true, 8);
    gemm(stream, M16, P_xa1, xn1a_1b, H1_16, 128, H1_32, 128, NN, 128, 128, 128, 108, false, 8);
    // ================= gin1 =================
    scatter_add<<<NE2, 64, 0, stream>>>(XF32, ei2, AGGX, NE2, 64, 37);
    add_to_h16<<<NN, 64, 0, stream>>>(XF32, AGGX, XA16, 64, 37);
    gemm(stream, XA16, P_x10, xn1_0b, M16, 128, nullptr, 0, NN, 64, 128, 64, 108, true, 8);
    gemm(stream, M16, P_x11, xn1_1b, H2_16, 128, H2_32, 128, NN, 128, 128, 128, 108, false, 8);
    // ================= attn1 + bn1 =================
    gemm(stream, H1_16, P_attn, attn_Wb, nullptr, 0, T1_32, 128, NN, 128, 128, 128, 108, false, 8);
    gemm(stream, H2_16, P_attn, attn_Wb, nullptr, 0, T2_32, 128, NN, 128, 128, 128, 108, false, 8);
    attn_score<<<(NN + 127) / 128, 128, 0, stream>>>(T1_32, T2_32, attn_v, ALPHA);
    attn_combine<<<NN, 128, 0, stream>>>(H1_32, H2_32, ALPHA, HATT);
    {
        int nch = (NN + 511) / 512, thr = nch * 108;
        bn_stats_t<float><<<(thr + 255) / 256, 256, 0, stream>>>(HATT, ST1, NN, 108, 128, 512);
    }
    bn_apply<<<NN, 128, 0, stream>>>(HATT, ST1, bn1_g, bn1_b, HBN32, HBN16, 108, 128, 1.0f / NN);

    // ================= conv2 (ba_gin, x_nn2) =================
    gemm(stream, HBN16, P_c2l, c2l_b, nullptr, 0, H32, 128, NN, 128, 128, 128, 108, false, 8);
    gemm(stream, EA16, P_c2e, c2e_b, nullptr, 0, EW32, 128, NE1, 32, 128, 32, 108, false, 8);
    ba_msg_agg<<<NE1, 128, 0, stream>>>(H32, EW32, ei1, AGG1D, NE1);
    add_to_h16<<<NN, 128, 0, stream>>>(H32, AGG1D, HS16, 128, 108);
    gemm(stream, HS16, P_x20, xn2_0b, M16, 128, nullptr, 0, NN, 128, 128, 128, 108, true, 8);
    gemm(stream, M16, P_x21, xn2_1b, H1_16, 128, H1_32, 128, NN, 128, 128, 128, 108, false, 8);
    // ================= gin2 (x_nn2) =================
    scatter_add<<<NE2, 128, 0, stream>>>(HBN32, ei2, AGGH, NE2, 128, 108);
    add_to_h16<<<NN, 128, 0, stream>>>(HBN32, AGGH, XAH16, 128, 108);
    gemm(stream, XAH16, P_x20, xn2_0b, M16, 128, nullptr, 0, NN, 128, 128, 128, 108, true, 8);
    gemm(stream, M16, P_x21, xn2_1b, H2_16, 128, H2_32, 128, NN, 128, 128, 128, 108, false, 8);
    // ================= attn2 + bn2 -> HFIN32 (persistent) =================
    gemm(stream, H1_16, P_attn, attn_Wb, nullptr, 0, T1_32, 128, NN, 128, 128, 128, 108, false, 8);
    gemm(stream, H2_16, P_attn, attn_Wb, nullptr, 0, T2_32, 128, NN, 128, 128, 128, 108, false, 8);
    attn_score<<<(NN + 127) / 128, 128, 0, stream>>>(T1_32, T2_32, attn_v, ALPHA);
    attn_combine<<<NN, 128, 0, stream>>>(H1_32, H2_32, ALPHA, HATT);
    {
        int nch = (NN + 511) / 512, thr = nch * 108;
        bn_stats_t<float><<<(thr + 255) / 256, 256, 0, stream>>>(HATT, ST2, NN, 108, 128, 512);
    }
    bn_apply<<<NN, 128, 0, stream>>>(HATT, ST2, bn2_g, bn2_b, HFIN32, nullptr, 108, 128, 1.0f / NN);

    // ================= edge phase (reuses node scratch) =================
    cur = scratch_mark;
    half_t* E22   = ah((size_t)NET * 64);
    half_t* E40a  = ah((size_t)NET * 64);
    half_t* E40   = ah((size_t)NET * 64);
    half_t* XE16  = ah((size_t)NET * 256);
    half_t* BIG_A = ah((size_t)NET * 512);
    half_t* BIG_B = ah((size_t)NET * 512);
    float*  OUT1  = af((size_t)NET * 64);

    // e = e_linear(ea): 11->22 relu ->40 relu ->40
    gemm(stream, EA16, P_el0, el0_b, E22, 64, nullptr, 0, NET, 32, 64, 32, 22, true, 4);
    gemm(stream, E22, P_el1, el1_b, E40a, 64, nullptr, 0, NET, 32, 64, 64, 40, true, 4);
    gemm(stream, E40a, P_el2, el2_b, E40, 64, nullptr, 0, NET, 64, 64, 64, 40, false, 4);
    // xe = [h[src] | h[dst] | e]
    build_xe<<<(NET + 127) / 128, 128, 0, stream>>>(HFIN32, E40, ei1, ei2, XE16);
    // xe_lin = relu(linear_e(xe)) ; bn3 ; segment-sum -> d_out[1024:]
    gemm(stream, XE16, P_le0, le0_b, BIG_A, 512, nullptr, 0, NET, 256, 512, 256, 512, true, 8);
    gemm(stream, BIG_A, P_le1, le1_b, BIG_B, 512, nullptr, 0, NET, 512, 512, 512, 512, true, 8);
    half_t* XE3 = BIG_A;   // reuse: 500000x256 f16
    gemm(stream, BIG_B, P_le2, le2_b, XE3, 256, nullptr, 0, NET, 512, 256, 512, 256, true, 8);
    {
        int nch = (NET + 511) / 512, thr = nch * 256;
        bn_stats_t<half_t><<<(thr + 255) / 256, 256, 0, stream>>>(XE3, ST3, NET, 256, 256, 512);
    }
    bn3_apply_seg<<<NET, 256, 0, stream>>>(XE3, ST3, bn3_g, bn3_b, ebatch, outp + NG, 1.0f / NET);
    // out = segment_sum(linear(xe)) -> d_out[0:1024]
    gemm(stream, XE16, P_l0, l0_b, BIG_A, 512, nullptr, 0, NET, 256, 512, 256, 512, true, 8);
    gemm(stream, BIG_A, P_l1, l1_b, BIG_B, 512, nullptr, 0, NET, 512, 512, 512, 512, true, 8);
    half_t* L3 = XE16;     // XE16 no longer needed after the GEMM above reads it
    gemm(stream, BIG_B, P_l2, l2_b, L3, 256, nullptr, 0, NET, 512, 256, 512, 256, true, 8);
    gemm(stream, L3, P_l3, l3_b, nullptr, 0, OUT1, 64, NET, 256, 64, 256, 1, false, 4);
    seg_out<<<(NET + 255) / 256, 256, 0, stream>>>(OUT1, ebatch, outp);
}
